// CrossAttention_16389595202302
// MI455X (gfx1250) — matrix operand order repaired and checked by reading, not compiled, not run on hardware
//
#include <hip/hip_runtime.h>

// ---------------------------------------------------------------------------
// CrossAttention for MI455X (gfx1250): bf16 WMMA pipeline, f32 accumulate,
// async-to-LDS double-buffered GEMM tile staging.
// b=8 n=4096 m=256 dim=1024 h=16 d=64. Keys padded 257 -> 272 (K) / 288 (V^T).
// ---------------------------------------------------------------------------

typedef unsigned short u16b;
typedef __attribute__((ext_vector_type(16))) __bf16 v16bf;
typedef __attribute__((ext_vector_type(8)))  float  v8f;

union Frag { v16bf v; uint4 q[2]; unsigned int u[8]; };

#define BATCH 8
#define NQ    4096
#define MKEY  256
#define DIM   1024
#define HEADS 16
#define DHEAD 64
#define KPAD  272   // 1 null + 256 keys + 15 zero pad (17 tiles of 16)
#define VPAD  288   // 9 WMMA k-chunks of 32
#define PSTRIDE 296 // LDS row stride for P (x2 bytes = 592 = 16B aligned)
#define MAXNEG (-3.402823466e38f)

__device__ __forceinline__ u16b f2bf(float f) {
  union { float f; unsigned u; } c; c.f = f;
  unsigned r = c.u + 0x7FFFu + ((c.u >> 16) & 1u);
  return (u16b)(r >> 16);
}

__device__ __forceinline__ v8f wmma_bf16(const Frag& a, const Frag& b, v8f c) {
  return __builtin_amdgcn_wmma_f32_16x16x32_bf16(
      false, a.v, false, b.v, (short)0, c, false, false);
}

// 16-byte global -> LDS async copy (ASYNCcnt-tracked, gfx1250).
__device__ __forceinline__ void async_cp16(unsigned lds_off, const u16b* g) {
  asm volatile("global_load_async_to_lds_b128 %0, %1, off"
               :
               : "v"(lds_off), "v"((unsigned long long)(uintptr_t)g)
               : "memory");
}
__device__ __forceinline__ void wait_async0() {
  asm volatile("s_wait_asynccnt 0x0" ::: "memory");
}

// ---------------------------------------------------------------------------
// Fused f32 -> bf16 cast + transpose: src [rows][cols] f32 -> dst [cols][rows]
// bf16. 64x64 LDS tiles. grid = (rows/64, cols/64), block = 256.
// ---------------------------------------------------------------------------
__global__ __launch_bounds__(256) void castT_k(const float* __restrict__ src,
                                               u16b* __restrict__ dst,
                                               int rows, int cols) {
  __shared__ u16b T[64 * 65];
  int tid = threadIdx.x;
  int r0 = blockIdx.x * 64, c0 = blockIdx.y * 64;
#pragma unroll
  for (int p = 0; p < 16; p++) {
    int i = p * 256 + tid, rr = i >> 6, cc = i & 63;
    T[rr * 65 + cc] = f2bf(src[(size_t)(r0 + rr) * cols + c0 + cc]);
  }
  __syncthreads();
#pragma unroll
  for (int p = 0; p < 16; p++) {
    int i = p * 256 + tid, cc = i >> 6, rr = i & 63;
    dst[(size_t)(c0 + cc) * rows + r0 + rr] = T[rr * 65 + cc];
  }
}

// ---------------------------------------------------------------------------
// LayerNorm over last dim (=1024), one block (256 thr) per row.
// BF16OUT=1: write bf16; BF16OUT=0: write f32 (in-place capable).
// ---------------------------------------------------------------------------
template <int BF16OUT>
__global__ __launch_bounds__(256) void ln_k(const float* __restrict__ x,
                                            const float* __restrict__ sc,
                                            const float* __restrict__ bi,
                                            void* __restrict__ out) {
  int row = blockIdx.x, tid = threadIdx.x;
  const float4* xr = (const float4*)(x + (size_t)row * DIM);
  float4 v = xr[tid];
  float s = v.x + v.y + v.z + v.w;
  float q = v.x * v.x + v.y * v.y + v.z * v.z + v.w * v.w;
#pragma unroll
  for (int off = 16; off > 0; off >>= 1) {
    s += __shfl_xor(s, off, 32);
    q += __shfl_xor(q, off, 32);
  }
  __shared__ float ss[8], sq[8];
  if ((tid & 31) == 0) { ss[tid >> 5] = s; sq[tid >> 5] = q; }
  __syncthreads();
  float S = 0.f, Q = 0.f;
#pragma unroll
  for (int i = 0; i < 8; i++) { S += ss[i]; Q += sq[i]; }
  float mu = S * (1.f / DIM);
  float var = Q * (1.f / DIM) - mu * mu;
  float rstd = rsqrtf(var + 1e-6f);
  int c = tid * 4;
  float y0 = (v.x - mu) * rstd * sc[c + 0] + bi[c + 0];
  float y1 = (v.y - mu) * rstd * sc[c + 1] + bi[c + 1];
  float y2 = (v.z - mu) * rstd * sc[c + 2] + bi[c + 2];
  float y3 = (v.w - mu) * rstd * sc[c + 3] + bi[c + 3];
  if (BF16OUT) {
    u16b* o = (u16b*)out + (size_t)row * DIM + c;
    uint2 p;
    p.x = (unsigned)f2bf(y0) | ((unsigned)f2bf(y1) << 16);
    p.y = (unsigned)f2bf(y2) | ((unsigned)f2bf(y3) << 16);
    *(uint2*)o = p;
  } else {
    float4* o = (float4*)((float*)out + (size_t)row * DIM);
    o[tid] = make_float4(y0, y1, y2, y3);
  }
}

// ---------------------------------------------------------------------------
// Null KV + zero padding init.
// ---------------------------------------------------------------------------
__global__ __launch_bounds__(64) void init_kv_k(const float* __restrict__ nkv,
                                                u16b* __restrict__ k,
                                                u16b* __restrict__ vt) {
  int bh = blockIdx.x, dd = threadIdx.x;
  k[((size_t)bh * KPAD + 0) * DHEAD + dd] = f2bf(nkv[dd]);
  for (int j = 257; j < KPAD; j++) k[((size_t)bh * KPAD + j) * DHEAD + dd] = 0;
  size_t vb = ((size_t)bh * DHEAD + dd) * VPAD;
  vt[vb + 0] = f2bf(nkv[DHEAD + dd]);
  for (int c = 257; c < VPAD; c++) vt[vb + c] = 0;
}

// ---------------------------------------------------------------------------
// bf16 WMMA GEMM: C[M,N] = A[M,K] * B[K,N], with B supplied transposed
// (Bt[N][K], K-major) so BOTH tiles stage via global_load_async_to_lds_b128.
// Block tile 128x128, 8 waves of 32x64 (2x4 WMMA tiles), double-buffered LDS
// (stride 40 -> every fragment is an aligned ds_load_b128, conflict-free).
// EPI: 0 = bf16 C, 1 = f32 C, 2 = KV scatter (K + V^T with +1 key offset).
// ---------------------------------------------------------------------------
template <int EPI>
__global__ __launch_bounds__(256) void gemm_k(
    const u16b* __restrict__ A, const u16b* __restrict__ Bt, int lda, int ldbt,
    int K, u16b* __restrict__ Cb, float* __restrict__ Cf, int ldc,
    u16b* __restrict__ kout, u16b* __restrict__ vtout) {
  __shared__ u16b As[2][128 * 40];
  __shared__ u16b Bs[2][128 * 40];
  int tid = threadIdx.x, lane = tid & 31, wv = tid >> 5;
  int wm = wv & 3, wn = wv >> 2;
  int hi = (lane >> 4) & 1, l15 = lane & 15;
  int row0 = blockIdx.x * 128, col0 = blockIdx.y * 128;

  unsigned aoff[2], boff[2];
#pragma unroll
  for (int b = 0; b < 2; b++) {
    aoff[b] = (unsigned)(uintptr_t)&As[b][0];
    boff[b] = (unsigned)(uintptr_t)&Bs[b][0];
  }

  v8f zf = {0.f, 0.f, 0.f, 0.f, 0.f, 0.f, 0.f, 0.f};
  v8f acc[2][4];
#pragma unroll
  for (int a = 0; a < 2; a++)
#pragma unroll
    for (int b = 0; b < 4; b++) acc[a][b] = zf;

  // Each thread copies 2 A-chunks + 2 B-chunks of 16B per k-step.
  auto issue = [&](int buf, int kk) {
#pragma unroll
    for (int p = 0; p < 2; p++) {
      int c = p * 256 + tid, row = c >> 2, ko = (c & 3) * 8;
      unsigned loff = (unsigned)((row * 40 + ko) * 2);
      async_cp16(aoff[buf] + loff, A + (size_t)(row0 + row) * lda + kk + ko);
      async_cp16(boff[buf] + loff, Bt + (size_t)(col0 + row) * ldbt + kk + ko);
    }
  };

  issue(0, 0);
  for (int k0 = 0; k0 < K; k0 += 32) {
    int buf = (k0 >> 5) & 1;
    wait_async0();       // my copies for `buf` done
    __syncthreads();     // everyone's copies for `buf` done; prev compute done
    if (k0 + 32 < K) issue(buf ^ 1, k0 + 32);  // prefetch overlaps compute

    const u16b* Ab = &As[buf][0];
    const u16b* Bb = &Bs[buf][0];
    Frag af[2];
#pragma unroll
    for (int mt = 0; mt < 2; mt++) {
      int r = wm * 32 + mt * 16 + l15;
      af[mt].q[0] = *(const uint4*)&Ab[r * 40 + (hi ? 8 : 0)];
      af[mt].q[1] = *(const uint4*)&Ab[r * 40 + (hi ? 24 : 16)];
    }
#pragma unroll
    for (int nt = 0; nt < 4; nt++) {
      int c = wn * 64 + nt * 16 + l15;
      Frag bf;
      bf.q[0] = *(const uint4*)&Bb[c * 40 + (hi ? 8 : 0)];
      bf.q[1] = *(const uint4*)&Bb[c * 40 + (hi ? 24 : 16)];
#pragma unroll
      for (int mt = 0; mt < 2; mt++) acc[mt][nt] = wmma_bf16(af[mt], bf, acc[mt][nt]);
    }
    __syncthreads();
  }

#pragma unroll
  for (int mt = 0; mt < 2; mt++)
#pragma unroll
    for (int nt = 0; nt < 4; nt++)
#pragma unroll
      for (int i = 0; i < 8; i++) {
        int gr = row0 + wm * 32 + mt * 16 + i + hi * 8;
        int gc = col0 + wn * 64 + nt * 16 + l15;
        float val = acc[mt][nt][i];
        if constexpr (EPI == 0) {
          Cb[(size_t)gr * ldc + gc] = f2bf(val);
        } else if constexpr (EPI == 1) {
          Cf[(size_t)gr * ldc + gc] = val;
        } else {
          int b = gr >> 8, j = (gr & 255) + 1;  // +1: slot 0 = null kv
          if (gc < DIM) {
            int h = gc >> 6, dd = gc & 63;
            kout[(((size_t)(b * HEADS + h) * KPAD) + j) * DHEAD + dd] = f2bf(val);
          } else {
            int c2 = gc - DIM, h = c2 >> 6, dd = c2 & 63;
            vtout[(((size_t)(b * HEADS + h) * DHEAD) + dd) * VPAD + j] = f2bf(val);
          }
        }
      }
}

// ---------------------------------------------------------------------------
// Attention: grid (64, b*h), block 64 (2 waves). Wave handles 32 queries.
// QK^T via 17 WMMA N-tiles -> f32 softmax (shuffle reductions) -> P bf16 in
// LDS -> PV via 9 k-chunk WMMAs. K is [b,h,272,64], V^T is [b,h,64,288].
// ---------------------------------------------------------------------------
__global__ __launch_bounds__(64) void attn_k(const u16b* __restrict__ qm,
                                             const u16b* __restrict__ km,
                                             const u16b* __restrict__ vtm,
                                             const unsigned char* __restrict__ mask,
                                             u16b* __restrict__ aout) {
  int tid = threadIdx.x, lane = tid & 31, w = tid >> 5;
  int hi = (lane >> 4) & 1, l15 = lane & 15;
  int bh = blockIdx.y, b = bh >> 4, h = bh & 15;
  int q0 = blockIdx.x * 64 + w * 32;

  __shared__ u16b P[2 * 32 * PSTRIDE];
  u16b* Pl = &P[w * 32 * PSTRIDE];
  for (int idx = lane; idx < 512; idx += 32) {  // zero pad cols 272..287
    int rr = idx >> 4, cc = 272 + (idx & 15);
    Pl[rr * PSTRIDE + cc] = 0;
  }

  const u16b* kb = km + (size_t)bh * KPAD * DHEAD;
  const u16b* vb = vtm + (size_t)bh * DHEAD * VPAD;
  v8f zf = {0.f, 0.f, 0.f, 0.f, 0.f, 0.f, 0.f, 0.f};

  for (int mt = 0; mt < 2; mt++) {
    int grow = b * NQ + q0 + mt * 16 + l15;
    const u16b* qr = qm + (size_t)grow * DIM + h * DHEAD;
    Frag qf[2];
#pragma unroll
    for (int kc = 0; kc < 2; kc++) {
      qf[kc].q[0] = *(const uint4*)&qr[kc * 32 + (hi ? 8 : 0)];
      qf[kc].q[1] = *(const uint4*)&qr[kc * 32 + (hi ? 24 : 16)];
    }
    v8f s[17];
#pragma unroll
    for (int t = 0; t < 17; t++) s[t] = zf;
#pragma unroll
    for (int t = 0; t < 17; t++) {
      const u16b* kr = kb + (size_t)(t * 16 + l15) * DHEAD;
#pragma unroll
      for (int kc = 0; kc < 2; kc++) {
        Frag bf;
        int ko = kc * 32 + (hi ? 8 : 0);
        bf.q[0] = *(const uint4*)&kr[ko];
        bf.q[1] = *(const uint4*)&kr[ko + 16];
        s[t] = wmma_bf16(qf[kc], bf, s[t]);
      }
      int j = t * 16 + l15;
      bool valid = (j == 0) || ((j <= MKEY) && (mask[b * MKEY + j - 1] != 0));
#pragma unroll
      for (int i = 0; i < 8; i++)
        s[t][i] = valid ? s[t][i] * 0.25f : MAXNEG;  // scale = heads^-0.5
    }
#pragma unroll
    for (int r = 0; r < 8; r++) {  // softmax rows (half-wave = 16 keys wide)
      float mx = MAXNEG;
#pragma unroll
      for (int t = 0; t < 17; t++) mx = fmaxf(mx, s[t][r]);
#pragma unroll
      for (int off = 1; off < 16; off <<= 1) mx = fmaxf(mx, __shfl_xor(mx, off, 32));
      float sum = 0.f;
#pragma unroll
      for (int t = 0; t < 17; t++) {
        float p = __expf(s[t][r] - mx);
        s[t][r] = p;
        sum += p;
      }
#pragma unroll
      for (int off = 1; off < 16; off <<= 1) sum += __shfl_xor(sum, off, 32);
      float inv = 1.0f / sum;
      int rl = mt * 16 + r + hi * 8;
#pragma unroll
      for (int t = 0; t < 17; t++)
        Pl[rl * PSTRIDE + t * 16 + l15] = f2bf(s[t][r] * inv);
    }
  }

  v8f o[2][4];
#pragma unroll
  for (int a = 0; a < 2; a++)
#pragma unroll
    for (int c = 0; c < 4; c++) o[a][c] = zf;
#pragma unroll
  for (int kc = 0; kc < 9; kc++) {
    Frag pf[2];
#pragma unroll
    for (int mt = 0; mt < 2; mt++) {
      int base = (mt * 16 + l15) * PSTRIDE + kc * 32;
      pf[mt].q[0] = *(const uint4*)&Pl[base + (hi ? 8 : 0)];
      pf[mt].q[1] = *(const uint4*)&Pl[base + (hi ? 24 : 16)];
    }
#pragma unroll
    for (int nt = 0; nt < 4; nt++) {
      const u16b* vr = vb + (size_t)(nt * 16 + l15) * VPAD;
      Frag vf;
      int ko = kc * 32 + (hi ? 8 : 0);
      vf.q[0] = *(const uint4*)&vr[ko];
      vf.q[1] = *(const uint4*)&vr[ko + 16];
#pragma unroll
      for (int mt = 0; mt < 2; mt++) o[mt][nt] = wmma_bf16(pf[mt], vf, o[mt][nt]);
    }
  }
#pragma unroll
  for (int mt = 0; mt < 2; mt++)
#pragma unroll
    for (int nt = 0; nt < 4; nt++)
#pragma unroll
      for (int i = 0; i < 8; i++) {
        int grow = b * NQ + q0 + mt * 16 + i + hi * 8;
        int col = h * DHEAD + nt * 16 + l15;
        aout[(size_t)grow * DIM + col] = f2bf(o[mt][nt][i]);
      }
}

// ---------------------------------------------------------------------------
// Host launch: castT -> LN -> q/kv GEMMs -> attention -> Wo GEMM -> LN(out)
// ---------------------------------------------------------------------------
extern "C" void kernel_launch(void* const* d_in, const int* in_sizes, int n_in,
                              void* d_out, int out_size, void* d_ws, size_t ws_size,
                              hipStream_t stream) {
  (void)in_sizes; (void)n_in; (void)out_size; (void)ws_size;
  const float* x      = (const float*)d_in[0];
  const float* ctx    = (const float*)d_in[1];
  const unsigned char* mask = (const unsigned char*)d_in[2];
  const float* ln1_s  = (const float*)d_in[3];
  const float* ln1_b  = (const float*)d_in[4];
  const float* lnc_s  = (const float*)d_in[5];
  const float* lnc_b  = (const float*)d_in[6];
  const float* Wq     = (const float*)d_in[7];
  const float* Wkv    = (const float*)d_in[8];
  const float* nullkv = (const float*)d_in[9];
  const float* Wo     = (const float*)d_in[10];
  const float* lno_s  = (const float*)d_in[11];
  const float* lno_b  = (const float*)d_in[12];

  char* ws = (char*)d_ws;
  const size_t rowsX = (size_t)BATCH * NQ;      // 32768
  const size_t rowsC = (size_t)BATCH * MKEY;    // 2048
  size_t off = 0;
  u16b* xn   = (u16b*)(ws + off); off += rowsX * DIM * 2;             // 64 MB (reused as attn_out)
  u16b* qb   = (u16b*)(ws + off); off += rowsX * DIM * 2;             // 64 MB
  u16b* cn   = (u16b*)(ws + off); off += rowsC * DIM * 2;             // 4 MB
  u16b* wqT  = (u16b*)(ws + off); off += (size_t)DIM * DIM * 2;       // 2 MB  [N][K]
  u16b* wkvT = (u16b*)(ws + off); off += (size_t)DIM * 2 * DIM * 2;   // 4 MB  [2N][K]
  u16b* woT  = (u16b*)(ws + off); off += (size_t)DIM * DIM * 2;       // 2 MB  [N][K]
  u16b* kb   = (u16b*)(ws + off); off += (size_t)BATCH * HEADS * KPAD * DHEAD * 2;
  u16b* vtb  = (u16b*)(ws + off); off += (size_t)BATCH * HEADS * DHEAD * VPAD * 2;
  u16b* attn_out = xn;  // xn dead after q GEMM

  // 1) weight cast + transpose -> bf16 [N][K]
  castT_k<<<dim3(16, 16), 256, 0, stream>>>(Wq, wqT, DIM, DIM);
  castT_k<<<dim3(16, 32), 256, 0, stream>>>(Wkv, wkvT, DIM, 2 * DIM);
  castT_k<<<dim3(16, 16), 256, 0, stream>>>(Wo, woT, DIM, DIM);
  // 2) LayerNorm + bf16 cast
  ln_k<1><<<(int)rowsX, 256, 0, stream>>>(x, ln1_s, ln1_b, xn);
  ln_k<1><<<(int)rowsC, 256, 0, stream>>>(ctx, lnc_s, lnc_b, cn);
  // 3) null kv + padding
  init_kv_k<<<BATCH * HEADS, 64, 0, stream>>>(nullkv, kb, vtb);
  // 4) q = xn @ Wq  (32768x1024x1024)
  gemm_k<0><<<dim3(256, 8), 256, 0, stream>>>(xn, wqT, DIM, DIM, DIM,
                                              qb, nullptr, DIM, nullptr, nullptr);
  // 5) kv = cn @ Wkv (2048x2048x1024) with K / V^T scatter
  gemm_k<2><<<dim3(16, 16), 256, 0, stream>>>(cn, wkvT, DIM, DIM, DIM,
                                              nullptr, nullptr, 0, kb, vtb);
  // 6) attention
  attn_k<<<dim3(NQ / 64, BATCH * HEADS), 64, 0, stream>>>(qb, kb, vtb, mask, attn_out);
  // 7) out = attn_out @ Wo -> f32 into d_out
  gemm_k<1><<<dim3(256, 8), 256, 0, stream>>>(attn_out, woT, DIM, DIM, DIM,
                                              nullptr, (float*)d_out, DIM, nullptr, nullptr);
  // 8) final LayerNorm, in place on d_out
  ln_k<0><<<(int)rowsX, 256, 0, stream>>>((const float*)d_out, lno_s, lno_b, d_out);
}
